// MotionAwareBlock_54022098649519
// MI455X (gfx1250) — compile-verified, matrix operand order checked by reading, not run on hardware
//
#include <hip/hip_runtime.h>
#include <hip/hip_bf16.h>

// ---------------------------------------------------------------------------
// MotionAwareBlock on gfx1250 (CDNA5): bf16 WMMA everywhere, fragment-layout
// pre-transposed intermediates, LDS-staged softmax.
// ---------------------------------------------------------------------------

typedef __attribute__((ext_vector_type(16))) __bf16 v16bf;
typedef __attribute__((ext_vector_type(8)))  __bf16 v8bf;
typedef __attribute__((ext_vector_type(8)))  float  v8f;

#define CAT16(lo, hi) __builtin_shufflevector((lo), (hi), 0,1,2,3,4,5,6,7,8,9,10,11,12,13,14,15)

namespace {
constexpr int B_    = 4;      // batch after folding SEQLEN
constexpr int SEQ_  = 4;
constexpr int C_    = 128;
constexpr int H_    = 112;
constexpr int W_    = 112;
constexpr int HW_   = H_ * W_;          // 12544
constexpr int NH_   = 8;
constexpr int NREG_ = 49;               // 7x7 regions
constexpr int TOK_  = 256;              // 16x16 tokens per region
constexpr int TOPK_ = 4;
constexpr size_t ELT = (size_t)B_ * HW_ * C_;   // 6,422,528
}

__device__ __forceinline__ v8f vzero_f() {
  v8f z;
#pragma unroll
  for (int i = 0; i < 8; ++i) z[i] = 0.0f;
  return z;
}
__device__ __forceinline__ v8bf vzero_bf() {
  v8bf z;
#pragma unroll
  for (int i = 0; i < 8; ++i) z[i] = (__bf16)0.0f;
  return z;
}
__device__ __forceinline__ v8f wmma_bf16(v16bf a, v16bf b, v8f c) {
  return __builtin_amdgcn_wmma_f32_16x16x32_bf16(false, a, false, b, (short)0, c, false, false);
}

// ---------------------------------------------------------------------------
// Prep: weights -> bf16 (layouts are already [o][c] row-major, flat copy)
// ---------------------------------------------------------------------------
__global__ void prep_weights(const float* __restrict__ wq, const float* __restrict__ wkv,
                             const float* __restrict__ wo,
                             __bf16* __restrict__ wq_bf, __bf16* __restrict__ wkv_bf,
                             __bf16* __restrict__ wo_bf) {
  int t = blockIdx.x * 256 + threadIdx.x;
  if (t < C_ * C_)      wq_bf[t]  = (__bf16)wq[t];
  if (t < 2 * C_ * C_)  wkv_bf[t] = (__bf16)wkv[t];
  if (t < C_ * C_)      wo_bf[t]  = (__bf16)wo[t];
}

// ---------------------------------------------------------------------------
// Prep: frame_now / frame_his (temporal exp weights), transposed to [b][p][c]
// bf16 so GEMM A-fragments are contiguous b128 loads.
// ---------------------------------------------------------------------------
__global__ void prep_frames(const float* __restrict__ x,
                            __bf16* __restrict__ nowT, __bf16* __restrict__ hisT) {
  size_t tid = (size_t)blockIdx.x * 256 + threadIdx.x;
  if (tid >= ELT) return;
  int p = (int)(tid % HW_);
  int c = (int)((tid / HW_) % C_);
  int b = (int)(tid / ((size_t)HW_ * C_));
  const size_t fs = (size_t)C_ * HW_;                       // frame stride
  const size_t s  = ((size_t)(b * SEQ_) * C_ + c) * HW_ + p;
  // weights = exp(-SEQLEN - t + 2) for t = 0,1,2
  float his = 0.13533528323661270f * x[s]
            + 0.04978706836786394f * x[s + fs]
            + 0.01831563888873418f * x[s + 2 * fs];
  float now = x[s + 3 * fs];
  size_t d = ((size_t)b * HW_ + p) * C_ + c;
  nowT[d] = (__bf16)now;
  hisT[d] = (__bf16)his;
}

// ---------------------------------------------------------------------------
// 1x1 conv as GEMM: one wave computes a 16(M=pixels) x 16(N=out-ch) tile,
// K=128 in 4 x v_wmma_f32_16x16x32_bf16.
//   mode 0: A=nowT,  W=wq   -> qT[b][reg][tok][c]
//   mode 1: A=hisT,  W=wkv  -> kT (o<128) / vT[b][reg][c][tok] + vf (o>=128)
//   mode 2: A=mT,    W=wo   -> outf (NCHW f32, final output)
// ---------------------------------------------------------------------------
__global__ void gemm_conv1x1(const __bf16* __restrict__ A, const __bf16* __restrict__ Bw,
                             const float* __restrict__ bias, int mode,
                             __bf16* __restrict__ qT, __bf16* __restrict__ kT,
                             __bf16* __restrict__ vT, float* __restrict__ vf,
                             float* __restrict__ outf) {
  const int lane = threadIdx.x & 31;
  const int wv   = threadIdx.x >> 5;
  const int l16  = lane & 15;
  const int sel  = lane >> 4;                 // 0: K 0-7/16-23, 1: K 8-15/24-31
  const int nt   = blockIdx.y * 4 + wv;
  const int o0   = nt * 16;
  const int gm   = blockIdx.x * 16;           // W%16==0 => tile stays in one region row
  const int b    = gm / HW_;
  const int p0   = gm % HW_;

  const __bf16* Ap = A  + (size_t)(gm + l16) * C_ + sel * 8;
  const __bf16* Bp = Bw + (size_t)(o0 + l16) * C_ + sel * 8;

  v8f acc = vzero_f();
#pragma unroll
  for (int kk = 0; kk < 4; ++kk) {
    const int k0 = kk * 32;
    v8bf alo = *(const v8bf*)(Ap + k0);
    v8bf ahi = *(const v8bf*)(Ap + k0 + 16);
    v8bf blo = *(const v8bf*)(Bp + k0);
    v8bf bhi = *(const v8bf*)(Bp + k0 + 16);
    __builtin_prefetch(Ap + k0 + 16 * C_, 0, 1);    // next M-tiles of A -> global_prefetch
    v16bf af = CAT16(alo, ahi);
    v16bf bf = CAT16(blo, bhi);
    acc = wmma_bf16(af, bf, acc);
  }
  const int   o  = o0 + l16;
  const float bv = bias[o];
#pragma unroll
  for (int j = 0; j < 8; ++j) acc[j] += bv;

  if (mode == 2) {
    float* dst = outf + ((size_t)b * C_ + o) * HW_ + p0 + sel * 8;
#pragma unroll
    for (int j = 0; j < 8; ++j) dst[j] = acc[j];
    return;
  }

  const int hh  = p0 / W_, ww = p0 % W_;       // ww % 16 == 0
  const int reg = (hh >> 4) * 7 + (ww >> 4);
  const int tb  = (hh & 15) * 16;              // token base (row within region)
  if (mode == 0 || o < C_) {
    __bf16* dst = (mode == 0) ? qT : kT;       // [b][reg][tok][c]
    size_t base = ((size_t)(b * NREG_ + reg) * TOK_ + tb + sel * 8) * C_ + o;
#pragma unroll
    for (int j = 0; j < 8; ++j) dst[base + (size_t)j * C_] = (__bf16)acc[j];
  } else {
    const int ov = o - C_;
    v8bf pk;
#pragma unroll
    for (int j = 0; j < 8; ++j) pk[j] = (__bf16)acc[j];
    // vT[b][reg][c][tok]: 8 consecutive tokens -> single b128 store
    *(v8bf*)(vT + ((size_t)(b * NREG_ + reg) * C_ + ov) * TOK_ + tb + sel * 8) = pk;
    float* vd = vf + ((size_t)b * C_ + ov) * HW_ + p0 + sel * 8;
#pragma unroll
    for (int j = 0; j < 8; ++j) vd[j] = acc[j];
  }
}

// ---------------------------------------------------------------------------
// Region means of q and k: qr/kr[b*49+reg][c]
// ---------------------------------------------------------------------------
__global__ void region_means(const __bf16* __restrict__ qT, const __bf16* __restrict__ kT,
                             float* __restrict__ qr, float* __restrict__ kr) {
  const int bi = blockIdx.x;                  // b*49+reg
  const int c  = threadIdx.x;                 // 0..127
  const __bf16* src = blockIdx.y ? kT : qT;
  float*        dst = blockIdx.y ? kr : qr;
  float s = 0.0f;
  for (int t = 0; t < TOK_; ++t) s += (float)src[((size_t)bi * TOK_ + t) * C_ + c];
  dst[(size_t)bi * C_ + c] = s * (1.0f / 256.0f);
}

// ---------------------------------------------------------------------------
// a_r = qr @ krT (49x49, K=128) + top-4 per row
// ---------------------------------------------------------------------------
__global__ void topk49(const float* __restrict__ qr, const float* __restrict__ kr,
                       int* __restrict__ idxo) {
  __shared__ float ar[NREG_ * NREG_];
  const int b   = blockIdx.x;
  const int tid = threadIdx.x;
  for (int e = tid; e < NREG_ * NREG_; e += 256) {
    int i = e / NREG_, j = e % NREG_;
    const float* qp = qr + ((size_t)b * NREG_ + i) * C_;
    const float* kp = kr + ((size_t)b * NREG_ + j) * C_;
    float s = 0.0f;
    for (int c = 0; c < C_; ++c) s += qp[c] * kp[c];
    ar[e] = s;
  }
  __syncthreads();
  if (tid < NREG_) {
    float row[NREG_];
    for (int j = 0; j < NREG_; ++j) row[j] = ar[tid * NREG_ + j];
    for (int k = 0; k < TOPK_; ++k) {
      int bi = 0; float bvv = row[0];
      for (int j = 1; j < NREG_; ++j) { if (row[j] > bvv) { bvv = row[j]; bi = j; } }
      idxo[((size_t)b * NREG_ + tid) * TOPK_ + k] = bi;
      row[bi] = -3.0e38f;
    }
  }
}

// ---------------------------------------------------------------------------
// Regional attention: one wave per (b,h,r, quarter-of-queries).
// QK: hd=16 zero-padded to K=32 bf16 WMMA; scores -> LDS bf16; two-pass
// softmax with shfl_xor reductions; AV: 32 WMMA k-steps, P re-read from LDS.
// ---------------------------------------------------------------------------
__global__ void attn_kernel(const __bf16* __restrict__ qT, const __bf16* __restrict__ kT,
                            const __bf16* __restrict__ vT, const int* __restrict__ idx,
                            float* __restrict__ aout) {
  __shared__ __attribute__((aligned(16))) __bf16 P[16 * 1024];
  __shared__ float sums[16];

  const int bid     = blockIdx.x;
  const int quarter = bid & 3;
  const int bhr     = bid >> 2;
  const int b   = bhr / (NH_ * NREG_);
  const int rem = bhr % (NH_ * NREG_);
  const int h   = rem / NREG_;
  const int r   = rem % NREG_;
  const int lane = threadIdx.x & 31;
  const int l16  = lane & 15;
  const int sel  = lane >> 4;

  int rg[TOPK_];
#pragma unroll
  for (int g = 0; g < TOPK_; ++g) rg[g] = idx[((size_t)b * NREG_ + r) * TOPK_ + g];

  const __bf16* qb = qT + ((size_t)(b * NREG_ + r) * TOK_) * C_ + h * 16;
  const float scale = 0.08838834764831845f;   // 128^-0.5
  const v8bf zb = vzero_bf();

  for (int mi = 0; mi < 4; ++mi) {
    const int mt = quarter * 4 + mi;          // 16-query tile index (0..15)
    // Q fragment: rows = tokens mt*16..+15, K = hd (16, zero-padded to 32)
    v8bf qlo = *(const v8bf*)(qb + (size_t)(mt * 16 + l16) * C_ + sel * 8);
    v16bf qa = CAT16(qlo, zb);

    // ---- QK^T: 64 n-tiles of 16 keys ----
    for (int g = 0; g < TOPK_; ++g) {
      const __bf16* kb = kT + ((size_t)(b * NREG_ + rg[g]) * TOK_) * C_ + h * 16;
#pragma unroll
      for (int ntl = 0; ntl < 16; ++ntl) {
        v8bf klo = *(const v8bf*)(kb + (size_t)(ntl * 16 + l16) * C_ + sel * 8);
        v16bf kfr = CAT16(klo, zb);
        v8f s = wmma_bf16(qa, kfr, vzero_f());
        const int nn = g * 256 + ntl * 16 + l16;
#pragma unroll
        for (int j = 0; j < 8; ++j) {
          int m = j + sel * 8;
          P[m * 1024 + nn] = (__bf16)(s[j] * scale);
        }
      }
    }
    __syncthreads();

    // ---- softmax over 1024 keys (lane l16 owns row l16; halves split cols) ----
    {
      const int row = l16, cb = sel * 512;
      float mx = -3.0e38f;
      for (int i = 0; i < 512; ++i) {
        float v = (float)P[row * 1024 + cb + i];
        mx = fmaxf(mx, v);
      }
      mx = fmaxf(mx, __shfl_xor(mx, 16, 32));
      float sm = 0.0f;
      for (int i = 0; i < 512; ++i) {
        float v = (float)P[row * 1024 + cb + i];
        float p = __expf(v - mx);
        P[row * 1024 + cb + i] = (__bf16)p;    // unnormalized prob, bf16
        sm += p;
      }
      sm += __shfl_xor(sm, 16, 32);
      if (sel == 0) sums[row] = sm;
    }
    __syncthreads();

    // ---- AV: K=1024 in 32 x WMMA(K=32); A=P from LDS, B=vT (contig tokens) ----
    v8f o = vzero_f();
    for (int kc = 0; kc < 32; ++kc) {
      const int g  = kc >> 3;
      const int t0 = (kc & 7) * 32;
      const __bf16* vb = vT + ((size_t)(b * NREG_ + rg[g]) * C_ + h * 16 + l16) * TOK_;
      v8bf plo = *(const v8bf*)(&P[l16 * 1024 + kc * 32 + sel * 8]);
      v8bf phi = *(const v8bf*)(&P[l16 * 1024 + kc * 32 + 16 + sel * 8]);
      v16bf pa = CAT16(plo, phi);
      v8bf vlo = *(const v8bf*)(vb + t0 + sel * 8);
      v8bf vhi = *(const v8bf*)(vb + t0 + 16 + sel * 8);
      v16bf vfr = CAT16(vlo, vhi);
      o = wmma_bf16(pa, vfr, o);
    }

    // ---- normalize + scatter to NCHW f32 ----
    const int hh0 = (r / 7) * 16 + mt;
    const int ww0 = (r % 7) * 16;
    size_t obase = ((size_t)b * C_ + h * 16 + l16) * HW_ + (size_t)hh0 * W_ + ww0 + sel * 8;
#pragma unroll
    for (int j = 0; j < 8; ++j) {
      int m = j + sel * 8;
      aout[obase + j] = o[j] / sums[m];
    }
    __syncthreads();   // protect P before next query tile
  }
}

// ---------------------------------------------------------------------------
// Depthwise 3x3(v) + bias + attention output, transposed to bf16 [b][p][c]
// (A-operand layout for the final 1x1 conv GEMM)
// ---------------------------------------------------------------------------
__global__ void dw_fuse(const float* __restrict__ aout, const float* __restrict__ vf,
                        const float* __restrict__ wl, const float* __restrict__ bl,
                        __bf16* __restrict__ mT) {
  size_t tid = (size_t)blockIdx.x * 256 + threadIdx.x;
  if (tid >= ELT) return;
  int p = (int)(tid % HW_);
  int c = (int)((tid / HW_) % C_);
  int b = (int)(tid / ((size_t)HW_ * C_));
  int hh = p / W_, ww = p % W_;
  const float* vp = vf + ((size_t)b * C_ + c) * HW_;
  float acc = bl[c] + aout[((size_t)b * C_ + c) * HW_ + p];
#pragma unroll
  for (int dy = -1; dy <= 1; ++dy) {
#pragma unroll
    for (int dx = -1; dx <= 1; ++dx) {
      int y = hh + dy, xx = ww + dx;
      if (y >= 0 && y < H_ && xx >= 0 && xx < W_)
        acc += wl[c * 9 + (dy + 1) * 3 + (dx + 1)] * vp[(size_t)y * W_ + xx];
    }
  }
  mT[((size_t)b * HW_ + p) * C_ + c] = (__bf16)acc;
}

// ---------------------------------------------------------------------------
extern "C" void kernel_launch(void* const* d_in, const int* in_sizes, int n_in,
                              void* d_out, int out_size, void* d_ws, size_t ws_size,
                              hipStream_t stream) {
  (void)in_sizes; (void)n_in; (void)out_size; (void)ws_size;
  const float* x   = (const float*)d_in[0];
  const float* wq  = (const float*)d_in[1];
  const float* bq  = (const float*)d_in[2];
  const float* wkv = (const float*)d_in[3];
  const float* bkv = (const float*)d_in[4];
  const float* wo  = (const float*)d_in[5];
  const float* bo  = (const float*)d_in[6];
  const float* wl  = (const float*)d_in[7];
  const float* bl  = (const float*)d_in[8];

  char* ws = (char*)d_ws;
  size_t off = 0;
  auto take = [&](size_t bytes) -> char* {
    char* p = ws + off;
    off = (off + bytes + 255) & ~(size_t)255;
    return p;
  };
  __bf16* nowT   = (__bf16*)take(ELT * 2);
  __bf16* hisT   = (__bf16*)take(ELT * 2);
  __bf16* qT     = (__bf16*)take(ELT * 2);
  __bf16* kT     = (__bf16*)take(ELT * 2);
  __bf16* vT     = (__bf16*)take(ELT * 2);
  __bf16* mT     = (__bf16*)take(ELT * 2);
  float*  vf     = (float*) take(ELT * 4);
  float*  aout   = (float*) take(ELT * 4);
  __bf16* wq_bf  = (__bf16*)take((size_t)C_ * C_ * 2);
  __bf16* wkv_bf = (__bf16*)take((size_t)2 * C_ * C_ * 2);
  __bf16* wo_bf  = (__bf16*)take((size_t)C_ * C_ * 2);
  float*  qr     = (float*) take((size_t)B_ * NREG_ * C_ * 4);
  float*  kr     = (float*) take((size_t)B_ * NREG_ * C_ * 4);
  int*    idxw   = (int*)   take((size_t)B_ * NREG_ * TOPK_ * 4);

  const int elt_blocks = (int)((ELT + 255) / 256);
  const int m_tiles    = (int)((size_t)B_ * HW_ / 16);      // 3136

  prep_weights<<<128, 256, 0, stream>>>(wq, wkv, wo, wq_bf, wkv_bf, wo_bf);
  prep_frames<<<elt_blocks, 256, 0, stream>>>(x, nowT, hisT);

  // q = conv1x1(frame_now, wq) ; N=128 -> 8 n-tiles, 4 waves/block
  gemm_conv1x1<<<dim3(m_tiles, 2), 128, 0, stream>>>(nowT, wq_bf, bq, 0,
                                                     qT, kT, vT, vf, nullptr);
  // kv = conv1x1(frame_his, wkv) ; N=256 -> 16 n-tiles
  gemm_conv1x1<<<dim3(m_tiles, 4), 128, 0, stream>>>(hisT, wkv_bf, bkv, 1,
                                                     qT, kT, vT, vf, nullptr);

  region_means<<<dim3(B_ * NREG_, 2), 128, 0, stream>>>(qT, kT, qr, kr);
  topk49<<<B_, 256, 0, stream>>>(qr, kr, idxw);

  attn_kernel<<<B_ * NH_ * NREG_ * 4, 32, 0, stream>>>(qT, kT, vT, idxw, aout);

  dw_fuse<<<elt_blocks, 256, 0, stream>>>(aout, vf, wl, bl, mT);

  // final conv1x1(out, wo) -> d_out (f32 NCHW)
  gemm_conv1x1<<<dim3(m_tiles, 2), 128, 0, stream>>>(mT, wo_bf, bo, 2,
                                                     nullptr, nullptr, nullptr, nullptr,
                                                     (float*)d_out);
}
